// BottleneckBlock_13426067768112
// MI455X (gfx1250) — matrix operand order repaired and checked by reading, not compiled, not run on hardware
//
#include <hip/hip_runtime.h>
#include <math.h>
#include <stdint.h>

typedef __attribute__((ext_vector_type(16))) __bf16 v16bf;
typedef __attribute__((ext_vector_type(8)))  float  v8f;
typedef __attribute__((ext_vector_type(4)))  int    v4i;

#define K_BINS 2048
#define EMB    64
#define NBATCH 8
#define TLEN   8192
#define NROWS  (NBATCH * TLEN)        /* 65536 */
#define XELEMS (NBATCH * EMB * TLEN)  /* 4194304 */

/* ws byte offsets (d_ws is 256-aligned from hipMalloc) */
#define WS_SCAL  0                              /* 5 floats: S, Sxx, fit, commit, masksum */
#define WS_KHI   64
#define WS_KLO   (WS_KHI + K_BINS * EMB * 2)
#define WS_KNORM (WS_KLO + K_BINS * EMB * 2)
#define WS_XL    (WS_KNORM + K_BINS * 4)
#define WS_MIND  (WS_XL + NROWS * 4)

#define STAGE_BINS 256
#define NSTAGES    (K_BINS / STAGE_BINS)        /* 8 */
#define TILES_PER_STAGE (STAGE_BINS / 16)       /* 16 */

/* ---- CDNA5 async global->LDS copy (ASYNCcnt-tracked), builtin if present ---- */
#if defined(__has_builtin)
#  if __has_builtin(__builtin_amdgcn_global_load_async_to_lds_b128)
#    define ASYNC_B128_BUILTIN 1
#  else
#    define ASYNC_B128_BUILTIN 0
#  endif
#  if __has_builtin(__builtin_amdgcn_s_wait_asynccnt)
#    define ASYNC_WAIT_BUILTIN 1
#  else
#    define ASYNC_WAIT_BUILTIN 0
#  endif
#else
#  define ASYNC_B128_BUILTIN 0
#  define ASYNC_WAIT_BUILTIN 0
#endif

__device__ __forceinline__ void async_copy_b128(const void* g, void* l) {
#if ASYNC_B128_BUILTIN
    /* signature: (global v4i*, local v4i*, imm offset, imm cpol) */
    __builtin_amdgcn_global_load_async_to_lds_b128(
        (__attribute__((address_space(1))) v4i*)(void*)g,
        (__attribute__((address_space(3))) v4i*)l, 0, 0);
#else
    unsigned loff = (unsigned)(uintptr_t)(__attribute__((address_space(3))) void*)l;
    unsigned long long ga = (unsigned long long)(uintptr_t)g;
    asm volatile("global_load_async_to_lds_b128 %0, %1, off"
                 :: "v"(loff), "v"(ga) : "memory");
#endif
}

__device__ __forceinline__ void wait_async0() {
#if ASYNC_WAIT_BUILTIN
    __builtin_amdgcn_s_wait_asynccnt(0);
#else
    asm volatile("s_wait_asynccnt 0x0" ::: "memory");
#endif
}

__global__ void zero_scal(float* scal) {
    if (threadIdx.x < 8) scal[threadIdx.x] = 0.f;
}

/* Split codebook into bf16 hi/lo, compute ||k_j||^2 */
__global__ void prep_k(const float* __restrict__ k, __bf16* __restrict__ khi,
                       __bf16* __restrict__ klo, float* __restrict__ knorm) {
    int bin = blockIdx.x * blockDim.x + threadIdx.x;
    if (bin >= K_BINS) return;
    float s = 0.f;
    for (int c = 0; c < EMB; ++c) {
        float f = k[bin * EMB + c];
        s += f * f;
        __bf16 h = (__bf16)f;
        khi[bin * EMB + c] = h;
        klo[bin * EMB + c] = (__bf16)(f - (float)h);
    }
    knorm[bin] = s;
}

/* Main VQ search: 8 waves/block, 16 rows per wave, double-buffered async
   LDS staging of the codebook.  dist' = ||k||^2 - 2 x.k ; row ||x||^2 added
   at the end. */
__global__ __launch_bounds__(256) void vq_main(
    const float* __restrict__ x,
    const __bf16* __restrict__ khi, const __bf16* __restrict__ klo,
    const float* __restrict__ knorm,
    float* __restrict__ out_xl, int* __restrict__ ws_xl,
    float* __restrict__ ws_mind)
{
    __shared__ __align__(32) __bf16 lds_hi[2][STAGE_BINS * EMB];  /* 2 x 32 KB */
    __shared__ __align__(32) __bf16 lds_lo[2][STAGE_BINS * EMB];  /* 2 x 32 KB */
    __shared__ __align__(16) float  lds_kn[2][STAGE_BINS];

    const int tid    = threadIdx.x;
    const int lane   = tid & 31;
    const int wave   = tid >> 5;
    const bool laneHi = (lane >= 16);
    const int mbase  = blockIdx.x * 128 + wave * 16;   /* 512 blocks * 128 rows */
    const int row_l  = mbase + (lane & 15);
    const int n      = row_l >> 13;                    /* /TLEN */
    const int t      = row_l & (TLEN - 1);

    /* ---- Build A tiles (16x32 bf16 per K-chunk) in contiguous-K layout (k = 16*half + elem) ---- */
    v16bf ahi[2], alo[2];
    float xnorm = 0.f;
#pragma unroll
    for (int c = 0; c < EMB; ++c) {
        float f = x[(size_t)(n * EMB + c) * TLEN + t];
        xnorm += f * f;
        __bf16 h = (__bf16)f;
        __bf16 l = (__bf16)(f - (float)h);
        const int cc    = c & 31;
        const int chunk = c >> 5;
        const int g     = cc >> 4;                        /* group of 16 K values */
        const int e     = cc & 15;                        /* element 0..15 */
        /* lanes 0-15 own K 0-15 ; lanes 16-31 own K 16-31 */
        if ((g != 0) == laneHi) {
            if (chunk == 0) { ahi[0][e] = h; alo[0][e] = l; }
            else            { ahi[1][e] = h; alo[1][e] = l; }
        }
    }

    /* async-stage one 256-bin chunk (hi: 32KB, lo: 32KB, kn: 1KB) */
    auto stage_load = [&](int stage, int buf) {
        const char* ghi = (const char*)(khi + (size_t)stage * STAGE_BINS * EMB);
        const char* glo = (const char*)(klo + (size_t)stage * STAGE_BINS * EMB);
        char* lhi = (char*)&lds_hi[buf][0];
        char* llo = (char*)&lds_lo[buf][0];
#pragma unroll
        for (int j = 0; j < 8; ++j) {
            const int off = (j * 256 + tid) * 16;
            async_copy_b128(ghi + off, lhi + off);
            async_copy_b128(glo + off, llo + off);
        }
        if (tid < 64) {   /* waves 0,1 fully active */
            const char* gkn = (const char*)(knorm + stage * STAGE_BINS);
            async_copy_b128(gkn + tid * 16, (char*)&lds_kn[buf][0] + tid * 16);
        }
    };

    float best[8];
    int   bidx[8];
#pragma unroll
    for (int r = 0; r < 8; ++r) { best[r] = 3.4e38f; bidx[r] = 0; }

    stage_load(0, 0);
    wait_async0();
    __syncthreads();

    for (int stage = 0; stage < NSTAGES; ++stage) {
        const int buf = stage & 1;
        if (stage + 1 < NSTAGES) stage_load(stage + 1, buf ^ 1);   /* overlap */

        const __bf16* Lhi = &lds_hi[buf][0];
        const __bf16* Llo = &lds_lo[buf][0];
        for (int nt = 0; nt < TILES_PER_STAGE; ++nt) {
            const int nl   = nt * 16 + (lane & 15);
            const int boff = nl * EMB + (laneHi ? 16 : 0);  /* B 32x16: lanes 0-15 K0-15, 16-31 K16-31 */
            v8f acc0 = {0.f, 0.f, 0.f, 0.f, 0.f, 0.f, 0.f, 0.f};
            v8f acc1 = {0.f, 0.f, 0.f, 0.f, 0.f, 0.f, 0.f, 0.f};
#pragma unroll
            for (int kc = 0; kc < 2; ++kc) {
                v16bf bh = *(const v16bf*)(Lhi + boff + kc * 32);
                v16bf bl = *(const v16bf*)(Llo + boff + kc * 32);
                acc0 = __builtin_amdgcn_wmma_f32_16x16x32_bf16(false, ahi[kc], false, bh, (short)0, acc0, false, false);
                acc1 = __builtin_amdgcn_wmma_f32_16x16x32_bf16(false, alo[kc], false, bh, (short)0, acc1, false, false);
                acc1 = __builtin_amdgcn_wmma_f32_16x16x32_bf16(false, ahi[kc], false, bl, (short)0, acc1, false, false);
            }
            const float kn = lds_kn[buf][nl];
            const int   Ng = stage * STAGE_BINS + nl;
#pragma unroll
            for (int r = 0; r < 8; ++r) {
                float d = fmaf(-2.f, acc0[r] + acc1[r], kn);
                if (d < best[r]) { best[r] = d; bidx[r] = Ng; }
            }
        }
        wait_async0();
        __syncthreads();
    }

    /* min-reduce over the 16 lanes of each half (lane = N, VGPR slot = M row) */
#pragma unroll
    for (int m = 1; m <= 8; m <<= 1) {
#pragma unroll
        for (int r = 0; r < 8; ++r) {
            float ov = __shfl_xor(best[r], m, 32);
            int   oi = __shfl_xor(bidx[r], m, 32);
            if (ov < best[r] || (ov == best[r] && oi < bidx[r])) { best[r] = ov; bidx[r] = oi; }
        }
    }

    /* writer lanes 0..7 -> rows mbase+0..7 ; lanes 16..23 -> rows mbase+8..15 */
    int   srcl   = laneHi ? (8 + (lane & 7)) : lane;
    float xn_sel = __shfl(xnorm, srcl, 32);
    float mval = best[0]; int midx = bidx[0];
#pragma unroll
    for (int r = 1; r < 8; ++r)
        if ((lane & 7) == r) { mval = best[r]; midx = bidx[r]; }

    if ((lane & 15) < 8) {
        int row = mbase + (laneHi ? 8 : 0) + (lane & 7);
        float md = xn_sel + mval;     /* full min distance == ||k[l]-x||^2 */
        out_xl[row]  = (float)midx;
        ws_xl[row]   = midx;
        ws_mind[row] = md;
    }
}

/* fit / commit / mask sums over rows */
__global__ void reduce_rows(const float* __restrict__ mind, const float* __restrict__ mask,
                            float* scal) {
    __shared__ float s0[256], s1[256], s2[256];
    int tid = threadIdx.x;
    float fit = 0.f, com = 0.f, msk = 0.f;
    for (int j = 0; j < 4; ++j) {
        int i = blockIdx.x * 1024 + j * 256 + tid;
        float md = mind[i], mm = mask[i];
        fit += md; com += mm * md; msk += mm;
    }
    s0[tid] = fit; s1[tid] = com; s2[tid] = msk;
    __syncthreads();
    for (int s = 128; s > 0; s >>= 1) {
        if (tid < s) { s0[tid] += s0[tid + s]; s1[tid] += s1[tid + s]; s2[tid] += s2[tid + s]; }
        __syncthreads();
    }
    if (tid == 0) {
        atomicAdd(&scal[2], s0[0]);
        atomicAdd(&scal[3], s1[0]);
        atomicAdd(&scal[4], s2[0]);
    }
}

/* sum(x) and sum(x^2) for prenorm */
__global__ void reduce_x(const float* __restrict__ x, float* scal) {
    __shared__ float s0[256], s1[256];
    int tid = threadIdx.x;
    float S = 0.f, Q = 0.f;
    size_t base = (size_t)blockIdx.x * 4096 + tid;
    for (int j = 0; j < 16; ++j) {
        float f = x[base + (size_t)j * 256];
        S += f; Q += f * f;
    }
    s0[tid] = S; s1[tid] = Q;
    __syncthreads();
    for (int s = 128; s > 0; s >>= 1) {
        if (tid < s) { s0[tid] += s0[tid + s]; s1[tid] += s1[tid + s]; }
        __syncthreads();
    }
    if (tid == 0) {
        atomicAdd(&scal[0], s0[0]);
        atomicAdd(&scal[1], s1[0]);
    }
}

/* x_d[n][c][t] = k[l][c] * mask ; coalesced along t */
__global__ void gather_out(const int* __restrict__ xl, const float* __restrict__ k,
                           const float* __restrict__ mask, float* __restrict__ out_xd) {
    int tid    = threadIdx.x;
    int strip  = blockIdx.x;        /* 16384 strips */
    int tchunk = strip & 31;
    int nc     = strip >> 5;        /* n*64 + c */
    int nn     = nc >> 6;
    int c      = nc & 63;
    int t      = tchunk * 256 + tid;
    int row    = nn * TLEN + t;
    int l      = xl[row];
    out_xd[(size_t)nc * TLEN + t] = k[l * EMB + c] * mask[row];
}

__global__ void finalize(const float* scal, float* out_sc) {
    if (threadIdx.x == 0 && blockIdx.x == 0) {
        float S = scal[0], Q = scal[1], fit = scal[2], com = scal[3], msk = scal[4];
        const float numel = (float)XELEMS;
        float var = (Q - S * S / numel) / numel;
        out_sc[0] = com / (msk * (float)EMB);   /* commit_loss */
        out_sc[1] = fit / (float)NROWS;         /* fit */
        out_sc[2] = sqrtf(fmaxf(var, 0.f));     /* prenorm */
    }
}

extern "C" void kernel_launch(void* const* d_in, const int* in_sizes, int n_in,
                              void* d_out, int out_size, void* d_ws, size_t ws_size,
                              hipStream_t stream) {
    const float* x    = (const float*)d_in[0];   /* [8, 64, 8192] */
    const float* mask = (const float*)d_in[1];   /* [8, 1, 8192]  */
    const float* k    = (const float*)d_in[2];   /* [2048, 64]    */

    char* ws = (char*)d_ws;
    float* scal    = (float*)(ws + WS_SCAL);
    __bf16* khi    = (__bf16*)(ws + WS_KHI);
    __bf16* klo    = (__bf16*)(ws + WS_KLO);
    float* knorm   = (float*)(ws + WS_KNORM);
    int*   ws_xl   = (int*)(ws + WS_XL);
    float* ws_mind = (float*)(ws + WS_MIND);

    float* out_xl = (float*)d_out;               /* NROWS floats (indices as float) */
    float* out_xd = out_xl + NROWS;              /* XELEMS floats */
    float* out_sc = out_xd + XELEMS;             /* 3 scalars */

    zero_scal<<<1, 32, 0, stream>>>(scal);
    prep_k<<<K_BINS / 256, 256, 0, stream>>>(k, khi, klo, knorm);
    vq_main<<<NROWS / 128, 256, 0, stream>>>(x, khi, klo, knorm, out_xl, ws_xl, ws_mind);
    reduce_rows<<<NROWS / 1024, 256, 0, stream>>>(ws_mind, mask, scal);
    reduce_x<<<XELEMS / 4096, 256, 0, stream>>>(x, scal);
    gather_out<<<NBATCH * EMB * (TLEN / 256), 256, 0, stream>>>(ws_xl, k, mask, out_xd);
    finalize<<<1, 32, 0, stream>>>(scal, out_sc);
}
